// OrthogonalBivectorBlock_72378788872516
// MI455X (gfx1250) — matrix operand order repaired and checked by reading, not compiled, not run on hardware
//
#include <hip/hip_runtime.h>
#include <hip/hip_bf16.h>
#include <math.h>

// ---------------------------------------------------------------------------
// Problem constants (match reference)
// ---------------------------------------------------------------------------
#define DD 512
#define LL 1024
#define BB 2
#define MM (BB * LL)          // 2048 rows
#define FEAT 64               // 32 key/query feature dims + 32 positional
#define PI_F 3.14159265358979323846f

typedef __attribute__((ext_vector_type(16))) _Float16 v16h;
typedef __attribute__((ext_vector_type(8)))  _Float16 v8h;
typedef __attribute__((ext_vector_type(8)))  float    v8f;

__device__ __forceinline__ v8f zero8() {
  v8f z = {0.f,0.f,0.f,0.f,0.f,0.f,0.f,0.f};
  return z;
}

__device__ __forceinline__ v16h cat16(v8h lo, v8h hi) {
  return __builtin_shufflevector(lo, hi, 0,1,2,3,4,5,6,7,8,9,10,11,12,13,14,15);
}

__device__ __forceinline__ float gelu_exact(float x) {
  return 0.5f * x * (1.0f + erff(x * 0.70710678118654752f));
}

// ---------------------------------------------------------------------------
// WMMA fragment loaders (wave32, V_WMMA_F32_16X16X32_F16) — vectorized.
// A 16x32 f16 (row-major tile, ld mult of 8 halves, 16B aligned):
//   lane m=lane&15, kh=(lane>>4)*8 -> runs [m,kh..kh+8) and [m,16+kh..+8)
// B 32x16 f16 from a column-major (NxK) tile:
//   lane n=lane&15, kh=(lane>>4)*8 -> runs [n,kh..kh+8) and [n,16+kh..+8)
// C/D 16x16 f32: VGPR r -> M = r + 8*(lane>>4), N = lane&15
// ---------------------------------------------------------------------------
__device__ __forceinline__ v16h frag_Av(const _Float16* p, int ld, int lane) {
  const int kh = (lane >> 4) * 8;
  const int m  = (lane & 15);
  v8h lo = *(const v8h*)(p + m * ld + kh);
  v8h hi = *(const v8h*)(p + m * ld + 16 + kh);
  return cat16(lo, hi);
}

__device__ __forceinline__ v16h frag_Bc(const _Float16* p, int ld, int lane) {
  const int kh = (lane >> 4) * 8;
  const int n  = (lane & 15);
  v8h lo = *(const v8h*)(p + n * ld + kh);
  v8h hi = *(const v8h*)(p + n * ld + 16 + kh);
  return cat16(lo, hi);
}

// ---------------------------------------------------------------------------
// f32 -> f16 conversion (row-major copy)
// ---------------------------------------------------------------------------
__global__ void cvt_kernel(const float* __restrict__ src,
                           _Float16* __restrict__ dst, int n) {
  int i = blockIdx.x * blockDim.x + threadIdx.x;
  if (i < n) dst[i] = (_Float16)src[i];
}

// f32 (K=512 x N) row-major  ->  f16 (N x K=512)
__global__ void cvtT_kernel(const float* __restrict__ src,
                            _Float16* __restrict__ dst, int Ndim, int total) {
  int i = blockIdx.x * blockDim.x + threadIdx.x;
  if (i < total) {
    int n = i >> 9;        // K = 512
    int k = i & 511;
    dst[(size_t)n * 512 + k] = (_Float16)src[(size_t)k * Ndim + n];
  }
}

// ---------------------------------------------------------------------------
// Tiled WMMA GEMM: C(MxN) = epilogue(A(MxK) @ B(KxN) + bias)
// A row-major f16; Bt = B transposed (NxK) f16.
// Block tile 128x128, 256 threads / 8 waves; wave grid 4(m) x 2(n);
// each wave: 2 m-tiles x 4 n-tiles = 8 WMMAs per 32-deep k-step.
// op: 0 = GELU -> f16 row-major, 1 = bias -> f16 TRANSPOSED out (outH[n*M+m]),
//     2 = bias + residual -> f32 row-major
// M,N multiples of 128; K multiple of 32.
// ---------------------------------------------------------------------------
enum { OP_GELU_F16 = 0, OP_BIAS_F16T = 1, OP_RES_F32 = 2 };

__global__ __launch_bounds__(256) void gemm_wmma_kernel(
    const _Float16* __restrict__ A, const _Float16* __restrict__ Bt,
    const float* __restrict__ bias, const float* __restrict__ res,
    _Float16* __restrict__ outH, float* __restrict__ outF,
    int M, int N, int K, int op)
{
  __shared__ alignas(16) _Float16 sA [128][40];   // 128x32 (+8 pad)
  __shared__ alignas(16) _Float16 sBt[128][40];   // 128(n) x 32(k) (+8 pad)

  const int tid  = threadIdx.x;
  const int wave = tid >> 5;
  const int lane = tid & 31;
  const int m0 = blockIdx.y * 128;
  const int n0 = blockIdx.x * 128;
  const int mg = (wave & 3) * 32;     // wave's 32-row m-group
  const int ng = (wave >> 2) * 64;    // wave's 64-col n-group

  // Fixed fill assignment: 128 rows x 4 v8h chunks = 512 chunks per array,
  // 2 per thread (rows fr and fr+64), straight-line, no predication.
  const int fr = tid >> 2;
  const int fo = (tid & 3) * 8;
  const _Float16* pA0 = A  + (size_t)(m0 + fr)      * K + fo;
  const _Float16* pA1 = A  + (size_t)(m0 + fr + 64) * K + fo;
  const _Float16* pB0 = Bt + (size_t)(n0 + fr)      * K + fo;
  const _Float16* pB1 = Bt + (size_t)(n0 + fr + 64) * K + fo;

  v8f acc[2][4];
  #pragma unroll
  for (int i = 0; i < 2; ++i)
    #pragma unroll
    for (int j = 0; j < 4; ++j) acc[i][j] = zero8();

  for (int k0 = 0; k0 < K; k0 += 32) {
    *(v8h*)&sA [fr][fo]      = *(const v8h*)(pA0 + k0);
    *(v8h*)&sA [fr + 64][fo] = *(const v8h*)(pA1 + k0);
    *(v8h*)&sBt[fr][fo]      = *(const v8h*)(pB0 + k0);
    *(v8h*)&sBt[fr + 64][fo] = *(const v8h*)(pB1 + k0);
    __syncthreads();

    v16h af0 = frag_Av(&sA[mg][0],      40, lane);
    v16h af1 = frag_Av(&sA[mg + 16][0], 40, lane);
    #pragma unroll
    for (int nt = 0; nt < 4; ++nt) {
      v16h bf = frag_Bc(&sBt[ng + nt * 16][0], 40, lane);
      acc[0][nt] = __builtin_amdgcn_wmma_f32_16x16x32_f16(
          false, af0, false, bf, (short)0, acc[0][nt], false, false);
      acc[1][nt] = __builtin_amdgcn_wmma_f32_16x16x32_f16(
          false, af1, false, bf, (short)0, acc[1][nt], false, false);
    }
    __syncthreads();
  }

  const int nc = lane & 15;
  #pragma unroll
  for (int mi = 0; mi < 2; ++mi) {
    const int mrow = m0 + mg + mi * 16 + 8 * (lane >> 4);
    if (op == OP_BIAS_F16T) {
      // transposed f16 output: each lane writes 8 consecutive m's as one b128
      #pragma unroll
      for (int nt = 0; nt < 4; ++nt) {
        int n = n0 + ng + nt * 16 + nc;
        float bv = bias[n];
        v8h pk;
        #pragma unroll
        for (int r = 0; r < 8; ++r) pk[r] = (_Float16)(acc[mi][nt][r] + bv);
        *(v8h*)&outH[(size_t)n * M + mrow] = pk;
      }
    } else {
      #pragma unroll
      for (int nt = 0; nt < 4; ++nt) {
        #pragma unroll
        for (int r = 0; r < 8; ++r) {
          int m = mrow + r;
          int n = n0 + ng + nt * 16 + nc;
          float v = acc[mi][nt][r] + bias[n];
          if (op == OP_GELU_F16) {
            outH[(size_t)m * N + n] = (_Float16)gelu_exact(v);
          } else {
            outF[(size_t)m * N + n] = v + res[(size_t)m * N + n];
          }
        }
      }
    }
  }
}

// ---------------------------------------------------------------------------
// Build Qfeat/Kfeat (64 dims each) per row r = b*L + l.
//   kp = tanh(Hk[r]·ke_w2 + ke_b2)·pi ; qp likewise ; gate = sigmoid(Hg[r]·gw2+gb2)
//   Kfeat = [cos kp (16), sin kp (16), cos ph (16), sin ph (16)]
//   Qfeat = [g·w_s·cos qp, g·w_s·sin qp, (1-g)·σ(pw)·cos ph, (1-g)·σ(pw)·sin ph]
// ---------------------------------------------------------------------------
__global__ __launch_bounds__(256) void phase_feat_kernel(
    const _Float16* __restrict__ Hk, const _Float16* __restrict__ Hq,
    const _Float16* __restrict__ Hg,
    const float* __restrict__ ke_w2, const float* __restrict__ ke_b2,
    const float* __restrict__ qe_w2, const float* __restrict__ qe_b2,
    const float* __restrict__ g_w2,  const float* __restrict__ g_b2,
    const float* __restrict__ set_w, const float* __restrict__ pos_w,
    const float* __restrict__ pos_f,
    _Float16* __restrict__ Qfeat, _Float16* __restrict__ Kfeat)
{
  __shared__ float sdots[32];
  __shared__ float sred[256];
  __shared__ float sscal[8];   // [0]=gate [1]=posw [2..5]=softmax(set_w)

  const int r = blockIdx.x;
  const int l = r & (LL - 1);
  const int tid = threadIdx.x, wave = tid >> 5, lane = tid & 31;

  // 32 dot products (16 key planes + 16 query planes), 4 per wave
  #pragma unroll
  for (int oo = 0; oo < 4; ++oo) {
    const int o = wave * 4 + oo;
    const _Float16* Arow = (o < 16) ? (Hk + (size_t)r * DD) : (Hq + (size_t)r * DD);
    const float*    W    = (o < 16) ? ke_w2 : qe_w2;
    const int p = o & 15;
    float s = 0.f;
    for (int d = lane; d < DD; d += 32) s += (float)Arow[d] * W[d * 16 + p];
    #pragma unroll
    for (int off = 16; off > 0; off >>= 1) s += __shfl_down(s, off, 32);
    if (lane == 0) sdots[o] = s;
  }

  // gate dot (256 dims) via block reduction
  sred[tid] = (float)Hg[(size_t)r * (DD / 2) + tid] * g_w2[tid];
  __syncthreads();
  for (int s = 128; s > 0; s >>= 1) {
    if (tid < s) sred[tid] += sred[tid + s];
    __syncthreads();
  }

  if (tid == 0) {
    float gate = 1.f / (1.f + expf(-(sred[0] + g_b2[0])));
    float posw = 1.f / (1.f + expf(-pos_w[0]));
    float mx = set_w[0];
    for (int i = 1; i < 4; ++i) mx = fmaxf(mx, set_w[i]);
    float e[4], tot = 0.f;
    for (int i = 0; i < 4; ++i) { e[i] = expf(set_w[i] - mx); tot += e[i]; }
    sscal[0] = gate; sscal[1] = posw;
    for (int i = 0; i < 4; ++i) sscal[2 + i] = e[i] / tot;
  }
  __syncthreads();

  const float gate = sscal[0], posw = sscal[1];
  _Float16* Kr = Kfeat + (size_t)r * FEAT;
  _Float16* Qr = Qfeat + (size_t)r * FEAT;
  if (tid < 16) {
    const int p = tid;
    float kph = tanhf(sdots[p] + ke_b2[p]) * PI_F;
    Kr[p]      = (_Float16)cosf(kph);
    Kr[16 + p] = (_Float16)sinf(kph);
    float qph = tanhf(sdots[16 + p] + qe_b2[p]) * PI_F;
    float ws = gate * sscal[2 + (p >> 2)];
    Qr[p]      = (_Float16)(ws * cosf(qph));
    Qr[16 + p] = (_Float16)(ws * sinf(qph));
  } else if (tid < 32) {
    const int f = tid - 16;
    float ph = (float)l * pos_f[f] * (2.0f * PI_F);
    float c = cosf(ph), s = sinf(ph);
    Kr[32 + f] = (_Float16)c;
    Kr[48 + f] = (_Float16)s;
    float a = (1.f - gate) * posw;
    Qr[32 + f] = (_Float16)(a * c);
    Qr[48 + f] = (_Float16)(a * s);
  }
}

// ---------------------------------------------------------------------------
// Causal un-normalized attention: Ret[b,l,:] = sum_{j<=l} (Qf[l]·Kf[j]) V[j,:]
// Vt is V transposed: Vt[d][m], m = b*L + j.
// One block per (32-query tile, batch); 8 waves.
// Waves 0-3 each build one 16x16 tile of S via WMMA (masked, f16 into LDS),
// then all 8 waves do S @ V (each a 16x128 slice of the 32x512 output).
// ---------------------------------------------------------------------------
__global__ __launch_bounds__(256) void attn_kernel(
    const _Float16* __restrict__ Qf, const _Float16* __restrict__ Kf,
    const _Float16* __restrict__ Vt, float* __restrict__ Ret)
{
  __shared__ alignas(16) _Float16 sQ[32][72];     // 32 x 64 (+8)
  __shared__ alignas(16) _Float16 sK[32][72];
  __shared__ alignas(16) _Float16 sS[32][40];     // 32 x 32 (+8)
  __shared__ alignas(16) _Float16 sVt[512][40];   // 512(d) x 32(j) (+8)

  const int tid  = threadIdx.x;
  const int wave = tid >> 5;
  const int lane = tid & 31;
  const int b  = blockIdx.y;
  const int q0 = blockIdx.x * 32;

  const _Float16* Qb = Qf + (size_t)b * LL * FEAT;
  const _Float16* Kb = Kf + (size_t)b * LL * FEAT;

  // load Q tile once: 32 rows x 64 halves = 256 v8h chunks
  {
    int r = tid >> 3, off = (tid & 7) * 8;
    *(v8h*)&sQ[r][off] = *(const v8h*)&Qb[(size_t)(q0 + r) * FEAT + off];
  }

  v8f acc[8];
  #pragma unroll
  for (int i = 0; i < 8; ++i) acc[i] = zero8();

  const int qm = (wave & 1) * 16;       // query sub-tile for this wave
  const int nb = (wave >> 1) * 128;     // 128-wide D slice for this wave

  for (int kt = 0; kt <= q0; kt += 32) {
    {
      int r = tid >> 3, off = (tid & 7) * 8;
      *(v8h*)&sK[r][off] = *(const v8h*)&Kb[(size_t)(kt + r) * FEAT + off];
    }
    // V tile: 512 d-rows x 32 keys = 2048 v8h chunks, 8 per thread
    #pragma unroll
    for (int c = 0; c < 8; ++c) {
      int d = (tid >> 2) + c * 64, off = (tid & 3) * 8;
      *(v8h*)&sVt[d][off] =
          *(const v8h*)&Vt[(size_t)d * MM + (size_t)b * LL + kt + off];
    }
    __syncthreads();

    if (wave < 4) {
      // wave w builds S tile (ms, ns): 2 WMMAs over the 64-dim features
      const int ms = (wave & 1) * 16;
      const int ns = (wave >> 1) * 16;
      v8f sacc = zero8();
      #pragma unroll
      for (int ks = 0; ks < 2; ++ks) {
        v16h aq = frag_Av(&sQ[ms][ks * 32], 72, lane);
        v16h bk = frag_Bc(&sK[ns][ks * 32], 72, lane);
        sacc = __builtin_amdgcn_wmma_f32_16x16x32_f16(
            false, aq, false, bk, (short)0, sacc, false, false);
      }
      const int srow = ms + 8 * (lane >> 4);
      #pragma unroll
      for (int rr = 0; rr < 8; ++rr) {
        int ml = srow + rr;
        int nl = ns + (lane & 15);
        float sv = sacc[rr];
        if (kt + nl > q0 + ml) sv = 0.f;   // causal mask (diagonal tile)
        sS[ml][nl] = (_Float16)sv;
      }
    }
    __syncthreads();

    v16h as = frag_Av(&sS[qm][0], 40, lane);
    #pragma unroll
    for (int nt = 0; nt < 8; ++nt) {
      v16h bv = frag_Bc(&sVt[nb + nt * 16][0], 40, lane);
      acc[nt] = __builtin_amdgcn_wmma_f32_16x16x32_f16(
          false, as, false, bv, (short)0, acc[nt], false, false);
    }
    __syncthreads();
  }

  const int mrow = qm + 8 * (lane >> 4);
  #pragma unroll
  for (int nt = 0; nt < 8; ++nt) {
    #pragma unroll
    for (int rr = 0; rr < 8; ++rr) {
      int m = q0 + mrow + rr;
      int n = nb + nt * 16 + (lane & 15);
      Ret[((size_t)b * LL + m) * DD + n] = acc[nt][rr];
    }
  }
}

// ---------------------------------------------------------------------------
// h = Ret / sqrt((l+1)*4); LayerNorm -> f16 (input to final GEMM)
// ---------------------------------------------------------------------------
__global__ __launch_bounds__(256) void scale_ln_kernel(
    const float* __restrict__ Ret, const float* __restrict__ ln_g,
    const float* __restrict__ ln_b, _Float16* __restrict__ Hln)
{
  __shared__ float red[256];
  __shared__ float stat[2];
  const int r = blockIdx.x, tid = threadIdx.x;
  const int l = r & (LL - 1);
  const float invn = rsqrtf((float)(l + 1) * 4.0f);
  const float* row = Ret + (size_t)r * DD;

  float h0 = row[tid] * invn;
  float h1 = row[tid + 256] * invn;

  red[tid] = h0 + h1;
  __syncthreads();
  for (int s = 128; s > 0; s >>= 1) {
    if (tid < s) red[tid] += red[tid + s];
    __syncthreads();
  }
  if (tid == 0) stat[0] = red[0] / (float)DD;
  __syncthreads();

  const float mu = stat[0];
  const float d0 = h0 - mu, d1 = h1 - mu;
  red[tid] = d0 * d0 + d1 * d1;
  __syncthreads();
  for (int s = 128; s > 0; s >>= 1) {
    if (tid < s) red[tid] += red[tid + s];
    __syncthreads();
  }
  if (tid == 0) stat[1] = red[0] / (float)DD;
  __syncthreads();

  const float rstd = rsqrtf(stat[1] + 1e-5f);
  Hln[(size_t)r * DD + tid]       = (_Float16)(d0 * rstd * ln_g[tid] + ln_b[tid]);
  Hln[(size_t)r * DD + tid + 256] = (_Float16)(d1 * rstd * ln_g[tid + 256] + ln_b[tid + 256]);
}

// ---------------------------------------------------------------------------
// Launch
// ---------------------------------------------------------------------------
extern "C" void kernel_launch(void* const* d_in, const int* in_sizes, int n_in,
                              void* d_out, int out_size, void* d_ws, size_t ws_size,
                              hipStream_t stream) {
  (void)in_sizes; (void)n_in; (void)out_size; (void)ws_size;

  const float* x     = (const float*)d_in[0];
  const float* ke_w1 = (const float*)d_in[1];
  const float* ke_b1 = (const float*)d_in[2];
  const float* ke_w2 = (const float*)d_in[3];
  const float* ke_b2 = (const float*)d_in[4];
  const float* qe_w1 = (const float*)d_in[5];
  const float* qe_b1 = (const float*)d_in[6];
  const float* qe_w2 = (const float*)d_in[7];
  const float* qe_b2 = (const float*)d_in[8];
  const float* v_w   = (const float*)d_in[9];
  const float* v_b   = (const float*)d_in[10];
  const float* ln_g  = (const float*)d_in[11];
  const float* ln_b  = (const float*)d_in[12];
  const float* out_w = (const float*)d_in[13];
  const float* out_b = (const float*)d_in[14];
  const float* set_w = (const float*)d_in[15];
  const float* pos_w = (const float*)d_in[16];
  const float* g_w1  = (const float*)d_in[17];
  const float* g_b1  = (const float*)d_in[18];
  const float* g_w2  = (const float*)d_in[19];
  const float* g_b2  = (const float*)d_in[20];
  const float* pos_f = (const float*)d_in[21];
  float* out = (float*)d_out;

  // Workspace bump allocator (256B aligned)
  char* p = (char*)d_ws;
  auto bump = [&](size_t bytes) -> char* {
    char* r = p;
    p += (bytes + 255) & ~(size_t)255;
    return r;
  };
  _Float16* xh    = (_Float16*)bump((size_t)MM * DD * 2);       // x, f16 row-major
  _Float16* kw1t  = (_Float16*)bump((size_t)DD * DD * 2);       // transposed weights
  _Float16* qw1t  = (_Float16*)bump((size_t)DD * DD * 2);
  _Float16* vwt   = (_Float16*)bump((size_t)DD * DD * 2);
  _Float16* owt   = (_Float16*)bump((size_t)DD * DD * 2);
  _Float16* gw1t  = (_Float16*)bump((size_t)(DD / 2) * DD * 2);
  _Float16* Hk    = (_Float16*)bump((size_t)MM * DD * 2);
  _Float16* Hq    = (_Float16*)bump((size_t)MM * DD * 2);
  _Float16* Hg    = (_Float16*)bump((size_t)MM * (DD / 2) * 2);
  _Float16* Vt    = (_Float16*)bump((size_t)DD * MM * 2);       // V transposed [d][m]
  _Float16* Qfeat = (_Float16*)bump((size_t)MM * FEAT * 2);
  _Float16* Kfeat = (_Float16*)bump((size_t)MM * FEAT * 2);
  float*    Ret   = (float*)   bump((size_t)MM * DD * 4);
  _Float16* Hln   = (_Float16*)bump((size_t)MM * DD * 2);

  cvt_kernel<<<dim3((MM * DD + 255) / 256), 256, 0, stream>>>(x, xh, MM * DD);
  auto cvtT = [&](const float* s, _Float16* d, int Ndim) {
    int total = 512 * Ndim;
    cvtT_kernel<<<dim3((total + 255) / 256), 256, 0, stream>>>(s, d, Ndim, total);
  };
  cvtT(ke_w1, kw1t, DD);
  cvtT(qe_w1, qw1t, DD);
  cvtT(v_w,   vwt,  DD);
  cvtT(out_w, owt,  DD);
  cvtT(g_w1,  gw1t, DD / 2);

  const dim3 gFull(DD / 128, MM / 128);        // N=512 -> (4,16)
  const dim3 gGate((DD / 2) / 128, MM / 128);  // N=256 -> (2,16)

  gemm_wmma_kernel<<<gFull, 256, 0, stream>>>(xh, kw1t, ke_b1, nullptr, Hk, nullptr,
                                              MM, DD, DD, OP_GELU_F16);
  gemm_wmma_kernel<<<gFull, 256, 0, stream>>>(xh, qw1t, qe_b1, nullptr, Hq, nullptr,
                                              MM, DD, DD, OP_GELU_F16);
  gemm_wmma_kernel<<<gFull, 256, 0, stream>>>(xh, vwt, v_b, nullptr, Vt, nullptr,
                                              MM, DD, DD, OP_BIAS_F16T);
  gemm_wmma_kernel<<<gGate, 256, 0, stream>>>(xh, gw1t, g_b1, nullptr, Hg, nullptr,
                                              MM, DD / 2, DD, OP_GELU_F16);

  phase_feat_kernel<<<MM, 256, 0, stream>>>(Hk, Hq, Hg, ke_w2, ke_b2, qe_w2, qe_b2,
                                            g_w2, g_b2, set_w, pos_w, pos_f,
                                            Qfeat, Kfeat);

  attn_kernel<<<dim3(LL / 32, BB), 256, 0, stream>>>(Qfeat, Kfeat, Vt, Ret);

  scale_ln_kernel<<<MM, 256, 0, stream>>>(Ret, ln_g, ln_b, Hln);

  gemm_wmma_kernel<<<gFull, 256, 0, stream>>>(Hln, owt, out_b, x, nullptr, out,
                                              MM, DD, DD, OP_RES_F32);
}
